// PinSAGEModel_88424786690459
// MI455X (gfx1250) — matrix operand order repaired and checked by reading, not compiled, not run on hardware
//
#include <hip/hip_runtime.h>

// ---------------------------------------------------------------------------
// CDNA5 (gfx1250, wave32) GraphSAGE forward.
//   - scatter aggregation: native f32 global atomics (L2-resident, 192MB L2)
//   - fused SAGE linear:  v_wmma_f32_16x16x32_bf16, K=256 ( [agg*deginv | x] )
//   - weights pre-packed into per-lane WMMA B-fragment layout (contiguous
//     32B v16bf load per lane per k-step)
// ---------------------------------------------------------------------------

typedef __attribute__((ext_vector_type(16))) __bf16 v16bf;
typedef __attribute__((ext_vector_type(8)))  float  v8f;

union ABFrag { v16bf v; unsigned int u[8]; };

__device__ __forceinline__ unsigned short f2bf(float f) {
  unsigned int b = __float_as_uint(f);
  b += 0x7FFFu + ((b >> 16) & 1u);          // round-to-nearest-even
  return (unsigned short)(b >> 16);
}

// ----------------------------- utility kernels -----------------------------

__global__ void zero_f32(float* __restrict__ p, long long n) {
  long long i = (long long)blockIdx.x * blockDim.x + threadIdx.x;
  long long stride = (long long)gridDim.x * blockDim.x;
  for (; i < n; i += stride) p[i] = 0.0f;
}

__global__ void degree_kernel(const long long* __restrict__ dst,
                              float* __restrict__ deg, long long E) {
  long long e = (long long)blockIdx.x * blockDim.x + threadIdx.x;
  if (e < E) {
    float* p = deg + dst[e];
    __hip_atomic_fetch_add(p, 1.0f, __ATOMIC_RELAXED, __HIP_MEMORY_SCOPE_AGENT);
  }
}

__global__ void deginv_kernel(float* __restrict__ deg, int n) {
  int i = blockIdx.x * blockDim.x + threadIdx.x;
  if (i < n) deg[i] = 1.0f / fmaxf(deg[i], 1.0f);
}

// One edge = 32 lanes x float4 = 128 contiguous floats (512B coalesced gather),
// then 4 relaxed agent-scope f32 atomic adds per lane into agg[dst].
__global__ void scatter_add128(const float* __restrict__ X,
                               const long long* __restrict__ src,
                               const long long* __restrict__ dst,
                               float* __restrict__ agg, long long E) {
  long long t = (long long)blockIdx.x * blockDim.x + threadIdx.x;
  if (t >= E * 32) return;
  long long e = t >> 5;
  int q = (int)(t & 31);
  long long s = src[e], d = dst[e];
  const float4 v = ((const float4*)(X + (size_t)s * 128))[q];
  float* o = agg + (size_t)d * 128 + (size_t)q * 4;
  __hip_atomic_fetch_add(o + 0, v.x, __ATOMIC_RELAXED, __HIP_MEMORY_SCOPE_AGENT);
  __hip_atomic_fetch_add(o + 1, v.y, __ATOMIC_RELAXED, __HIP_MEMORY_SCOPE_AGENT);
  __hip_atomic_fetch_add(o + 2, v.z, __ATOMIC_RELAXED, __HIP_MEMORY_SCOPE_AGENT);
  __hip_atomic_fetch_add(o + 3, v.w, __ATOMIC_RELAXED, __HIP_MEMORY_SCOPE_AGENT);
}

// Pack fp32 weight(s) [K x Ncols] (optionally concatenated Wa;Wb along K) into
// bf16 WMMA B-fragment order: dst[((ks*nct + ct)*32 + lane)*16 + e] where the
// element maps to K = ks*32 + (lane<16 ? 0:8) + (e<8 ? e : e+8), N = ct*16 + (lane&15).
__global__ void pack_weights_bf16(const float* __restrict__ Wa,
                                  const float* __restrict__ Wb, int kA,
                                  unsigned short* __restrict__ dst,
                                  int K, int Ncols, long long total) {
  long long idx = (long long)blockIdx.x * blockDim.x + threadIdx.x;
  if (idx >= total) return;
  int e    = (int)(idx & 15);
  int lane = (int)((idx >> 4) & 31);
  long long rest = idx >> 9;
  int nct = Ncols >> 4;
  int ct = (int)(rest % nct);
  int ks = (int)(rest / nct);
  int k = ks * 32 + ((lane < 16) ? 0 : 8) + ((e < 8) ? e : e + 8);
  int n = ct * 16 + (lane & 15);
  float v = (k < kA) ? Wa[(long long)k * Ncols + n]
                     : Wb[(long long)(k - kA) * Ncols + n];
  dst[idx] = f2bf(v);
}

// --------------------------- fused SAGE GEMM -------------------------------
// out[16 rows x ncols] per block.  A = [A0*deginv | A1] staged to LDS as bf16
// (K=256), or A = A0 alone (K=128).  One wave per 16-col tile.
// Epilogue fuses bias, optional residual add, optional ReLU.  Writing out==A1
// in-place is safe: all A reads are staged to LDS before stores, and the
// residual read/write is the same address in the same lane.
__global__ __launch_bounds__(256) void sage_gemm_wmma(
    const float* __restrict__ A0, const float* __restrict__ A1,
    const float* __restrict__ deginv, const unsigned short* __restrict__ Wp,
    const float* __restrict__ bias, const float* __restrict__ addIn,
    float* __restrict__ out, int nrows, int K, int ncols,
    int scaleA0, int doRelu) {
  __shared__ unsigned short sA[16 * 256];  // 8 KB of the 320 KB/WGP LDS

  const int row0 = blockIdx.x * 16;
  const int tid  = threadIdx.x;
  const int kshift = (K == 256) ? 8 : 7;
  const int kmask  = K - 1;

  // Stage A tile (16 x K) into LDS as bf16, applying deg_inv to the agg half.
  for (int i = tid; i < 16 * K; i += blockDim.x) {
    int r = i >> kshift;
    int k = i & kmask;
    int row = row0 + r;
    float v = 0.0f;
    if (row < nrows) {
      if (A1 == nullptr || k < 128) {
        v = A0[(size_t)row * 128 + k];
        if (scaleA0) v *= deginv[row];
      } else {
        v = A1[(size_t)row * 128 + (k - 128)];
      }
    }
    sA[i] = f2bf(v);
  }
  __syncthreads();

  const int wave = tid >> 5;
  const int lane = tid & 31;
  const int nct  = ncols >> 4;

  // A fragment addressing (16-bit A 16x32 layout, wave32):
  //   lane<16: M=lane,    K pairs {0..7, 16..23}
  //   lane>=16: M=lane-16, K pairs {8..15, 24..31}
  const int m = lane & 15;
  const int koffu = (lane < 16) ? 0 : 4;  // offset in uint (2 bf16) units
  const unsigned int* rowp = (const unsigned int*)(sA + m * K);
  const v16bf* wp = (const v16bf*)Wp;

  v8f acc = {};
  const int ksteps = K >> 5;
  for (int ks = 0; ks < ksteps; ++ks) {
    ABFrag a, b;
    const int base = ks * 16;  // ks*32 bf16 = ks*16 uints
#pragma unroll
    for (int v = 0; v < 4; ++v) a.u[v]     = rowp[base + koffu + v];
#pragma unroll
    for (int v = 0; v < 4; ++v) a.u[4 + v] = rowp[base + 8 + koffu + v];
    b.v = wp[((size_t)ks * nct + wave) * 32 + lane];
    acc = __builtin_amdgcn_wmma_f32_16x16x32_bf16(
        /*neg_a=*/false, a.v, /*neg_b=*/false, b.v,
        /*c_mod=*/(short)0, acc, /*reuse_a=*/false, /*reuse_b=*/false);
  }

  // C/D layout: VGPR v -> M = v + (lane<16 ? 0 : 8), N = lane&15.
  const int n   = lane & 15;
  const int col = wave * 16 + n;
  const int mb  = (lane < 16) ? 0 : 8;
  const float bc = bias[col];
#pragma unroll
  for (int v = 0; v < 8; ++v) {
    int row = row0 + v + mb;
    if (row < nrows) {
      float val = acc[v] + bc;
      if (addIn) val += addIn[(size_t)row * ncols + col];
      if (doRelu) val = fmaxf(val, 0.0f);
      out[(size_t)row * ncols + col] = val;
    }
  }
}

// ------------------------------- launcher ----------------------------------

extern "C" void kernel_launch(void* const* d_in, const int* in_sizes, int n_in,
                              void* d_out, int out_size, void* d_ws, size_t ws_size,
                              hipStream_t stream) {
  const float*     x    = (const float*)d_in[0];
  const long long* ei   = (const long long*)d_in[1];  // int64 [2, E]
  const float*     W1l  = (const float*)d_in[2];
  const float*     b1   = (const float*)d_in[3];
  const float*     W1r  = (const float*)d_in[4];
  const float*     W2l  = (const float*)d_in[5];
  const float*     b2   = (const float*)d_in[6];
  const float*     W2r  = (const float*)d_in[7];
  const float*     Wlin = (const float*)d_in[8];
  const float*     blin = (const float*)d_in[9];
  float*           out  = (float*)d_out;

  const long long N = in_sizes[0] / 128;
  const long long E = in_sizes[1] / 2;
  const int OUTC    = in_sizes[9];

  // Workspace carve-out (256B aligned)
  char* ws = (char*)d_ws;
  size_t off = 0;
  auto carve = [&](size_t bytes) -> void* {
    void* p = (void*)(ws + off);
    off = (off + bytes + 255) & ~(size_t)255;
    return p;
  };
  float* deg = (float*)carve((size_t)N * 4);           // degree -> deg_inv
  float* agg = (float*)carve((size_t)N * 128 * 4);     // neighbor sums
  float* h1  = (float*)carve((size_t)N * 128 * 4);     // hidden (reused in place)
  unsigned short* Wp1 = (unsigned short*)carve(256 * 128 * 2);
  unsigned short* Wp2 = (unsigned short*)carve(256 * 128 * 2);
  unsigned short* WpL = (unsigned short*)carve((size_t)128 * OUTC * 2);

  const long long* src = ei;
  const long long* dst = ei + E;

  // 1) zero degree + agg
  zero_f32<<<2048, 256, 0, stream>>>(deg, N);
  zero_f32<<<4096, 256, 0, stream>>>(agg, N * 128);

  // 2) pack bf16 weight fragments (tiny, once per launch)
  {
    long long t1 = 256LL * 128;
    int g1 = (int)((t1 + 255) / 256);
    pack_weights_bf16<<<g1, 256, 0, stream>>>(W1l, W1r, 128, Wp1, 256, 128, t1);
    pack_weights_bf16<<<g1, 256, 0, stream>>>(W2l, W2r, 128, Wp2, 256, 128, t1);
    long long t2 = 128LL * OUTC;
    pack_weights_bf16<<<(int)((t2 + 255) / 256), 256, 0, stream>>>(
        Wlin, nullptr, 128, WpL, 128, OUTC, t2);
  }

  // 3) degree -> deg_inv
  degree_kernel<<<(int)((E + 255) / 256), 256, 0, stream>>>(dst, deg, E);
  deginv_kernel<<<(int)((N + 255) / 256), 256, 0, stream>>>(deg, (int)N);

  const long long T = E * 32;
  const int sgrid = (int)((T + 255) / 256);
  const int gblk  = (int)((N + 15) / 16);

  // 4) layer 1: scatter x, fused GEMM -> h1 = relu([agg*dinv | x] @ [W1l;W1r] + b1)
  scatter_add128<<<sgrid, 256, 0, stream>>>(x, src, dst, agg, E);
  sage_gemm_wmma<<<gblk, 256, 0, stream>>>(agg, x, deg, Wp1, b1, nullptr, h1,
                                           (int)N, 256, 128, 1, 1);

  // 5) layer 2: re-zero agg, scatter h1, fused GEMM with residual,
  //    h1 <- relu(h1 + [agg*dinv | h1] @ [W2l;W2r] + b2)   (in-place safe)
  zero_f32<<<4096, 256, 0, stream>>>(agg, N * 128);
  scatter_add128<<<sgrid, 256, 0, stream>>>(h1, src, dst, agg, E);
  sage_gemm_wmma<<<gblk, 256, 0, stream>>>(agg, h1, deg, Wp2, b2, h1, h1,
                                           (int)N, 256, 128, 1, 1);

  // 6) linear head: out = h1 @ Wlin + blin   (K=128, OUT cols, 4 waves/block)
  sage_gemm_wmma<<<gblk, 32 * (OUTC / 16), 0, stream>>>(
      h1, nullptr, nullptr, WpL, blin, nullptr, out, (int)N, 128, OUTC, 0, 0);
}
